// ConvCapsuleLayer_29154238005457
// MI455X (gfx1250) — matrix operand order repaired and checked by reading, not compiled, not run on hardware
//
#include <hip/hip_runtime.h>
#include <hip/hip_bf16.h>

// ---------------------------------------------------------------------------
// ConvCapsuleLayer, fully fused for MI455X (gfx1250, wave32).
//
// Conv = implicit GEMM (M=16 px, N=256, K=800) on v_wmma_f32_16x16x32_f16,
// fused with 3 dynamic-routing iterations (256 MB of votes never touch HBM).
// Weights repacked once to f16 [N=256][K=800] in d_ws (410 KB, L2-resident).
//
// The compiler's scheduler insists on a single A buffer and near-zero B
// prefetch distance (ds_load x2 -> s_wait 0 -> wmma; global pair -> wait 0
// every 4 wmmas), so the GEMM inner loop is hand-pipelined with inline asm:
//   - B: global_load_b128 pairs into a 4-slot ring, issued 3 K-chunks
//     (12 WMMAs) ahead of use; readiness via manual "s_wait_loadcnt N"
//     (VMEM returns in order, so threshold waits are exact and extra
//     compiler loads only make them more conservative).
//   - A: all 8 ds_load_b128 of a chunk issue back-to-back, then drain with
//     "s_wait_dscnt 6/4/2/0" interleaved between the 4 WMMAs.
// Wait-asm carries the fragment as "+v" so it pins load -> wait -> wmma.
// ---------------------------------------------------------------------------

typedef __attribute__((ext_vector_type(16))) _Float16 v16h;
typedef __attribute__((ext_vector_type(8)))  float    v8f;

#define H_IMG 128
#define W_IMG 128
#define B_SZ  4
#define CI    4
#define AI    32
#define CO    8
#define AT    32
#define NCH   256            // CO*AT
#define KTOT  800            // 5*5*AI
#define PSTR  40             // padded atom stride in LDS patch (bank spread)

#define PATCH_BYTES (CI*5*20*PSTR*2)   // 32000
#define VOTES_BYTES (16*NCH*CI*4)      // 65536  ([px][256][ci] f32)
#define ROUTE_BYTES (16*32*4)          // 2048
#define ACT_BYTES   (16*NCH*4)         // 16384
#define SMEM_BYTES  (PATCH_BYTES + VOTES_BYTES + ROUTE_BYTES + ACT_BYTES)  // 115968

// --- one-time weight repack: W[kh][kw][ai][n] f32  ->  Wh[n][k] f16 --------
__global__ void repack_w_kernel(const float* __restrict__ W,
                                _Float16* __restrict__ Wh) {
  int t = blockIdx.x * blockDim.x + threadIdx.x;
  if (t >= NCH * KTOT) return;
  int n = t / KTOT;
  int k = t - n * KTOT;          // k = (kh*5+kw)*32 + ai
  Wh[t] = (_Float16)W[k * NCH + n];
}

// ---- manual-pipeline primitives (asm: scheduler cannot sink/merge) --------
template <int OFF>
__device__ __forceinline__ void glb_load_frag(v16h& r, const _Float16* p) {
  asm volatile("global_load_b128 %0, %2, off offset:%c3\n\t"
               "global_load_b128 %1, %2, off offset:%c4"
               : "=v"(((uint4*)&r)[0]), "=v"(((uint4*)&r)[1])
               : "v"(p), "i"(OFF), "i"(OFF + 16));
}
template <int OFF>
__device__ __forceinline__ void ds_load_frag(v16h& r, unsigned addr) {
  asm volatile("ds_load_b128 %0, %2 offset:%c3\n\t"
               "ds_load_b128 %1, %2 offset:%c4"
               : "=v"(((uint4*)&r)[0]), "=v"(((uint4*)&r)[1])
               : "v"(addr), "i"(OFF), "i"(OFF + 16));
}
template <int N>
__device__ __forceinline__ void wait_ld(v16h& f) {   // pins load->wait->use
  asm volatile("s_wait_loadcnt %c1" : "+v"(f) : "i"(N));
}
template <int N>
__device__ __forceinline__ void wait_ds(v16h& f) {
  asm volatile("s_wait_dscnt %c1" : "+v"(f) : "i"(N));
}

#define WMMA(acc, a, b)                                              \
  acc = __builtin_amdgcn_wmma_f32_16x16x32_f16(false, (a), false, (b), \
                                               (short)0, (acc), false, false)

// LDS byte offset of A fragment for chunk kc, input capsule ci
#define AOFF(kc, ci) ((ci) * 8000 + ((kc) / 5) * 3200 + ((kc) % 5) * 80)
// B wait threshold: pairs kc..min(kc+3,24) in flight; pair kc done at <=N
#define WBIMM(kc) ((kc) <= 21 ? 6 : ((24 - (kc)) * 2))

// one K-chunk: issue B(kc+3), issue 8 A loads, drain incrementally, 4 WMMAs
#define CAPS_STEP(kc)                                                   \
  do {                                                                  \
    if constexpr ((kc) + 3 <= 24)                                       \
      glb_load_frag<((kc) + 3) * 64>(Bring[((kc) + 3) & 3], bbase);     \
    ds_load_frag<AOFF(kc, 0)>(A0, abase);                               \
    ds_load_frag<AOFF(kc, 1)>(A1, abase);                               \
    ds_load_frag<AOFF(kc, 2)>(A2, abase);                               \
    ds_load_frag<AOFF(kc, 3)>(A3, abase);                               \
    wait_ld<WBIMM(kc)>(Bring[(kc) & 3]);                                \
    wait_ds<6>(A0);                                                     \
    WMMA(acc0, A0, Bring[(kc) & 3]);                                    \
    wait_ds<4>(A1);                                                     \
    WMMA(acc1, A1, Bring[(kc) & 3]);                                    \
    wait_ds<2>(A2);                                                     \
    WMMA(acc2, A2, Bring[(kc) & 3]);                                    \
    wait_ds<0>(A3);                                                     \
    WMMA(acc3, A3, Bring[(kc) & 3]);                                    \
  } while (0)

// --- fused conv (WMMA implicit GEMM) + dynamic routing ---------------------
__global__ __launch_bounds__(512, 1)
void capsconv_fused_kernel(const float* __restrict__ x,
                           const _Float16* __restrict__ Wh,
                           const float* __restrict__ bias,
                           float* __restrict__ out) {
  extern __shared__ char smem[];
  _Float16* patch = (_Float16*)smem;
  float*    vlds  = (float*)(smem + PATCH_BYTES);   // [16 px][256 n][4 ci]
  float*    rlds  = (float*)(smem + PATCH_BYTES + VOTES_BYTES);
  float*    alds  = (float*)(smem + PATCH_BYTES + VOTES_BYTES + ROUTE_BYTES);

  const int w0   = blockIdx.x * 16;   // 16-pixel row tile
  const int h    = blockIdx.y;
  const int b    = blockIdx.z;
  const int tid  = threadIdx.x;
  const int wv   = tid >> 5;          // wave id = N-tile = pixel
  const int lane = tid & 31;

  // ---- stage zero-padded input halo to LDS, f32 -> f16 (4 atoms/step) ----
  for (int t = tid; t < CI * 5 * 20 * 8; t += 512) {
    int ai4 = t & 7;
    int c   = (t >> 3) % 20;
    int r   = ((t >> 3) / 20) % 5;
    int ci  = t / (8 * 20 * 5);
    int gh = h + r - 2, gw = w0 + c - 2;
    float4 v = make_float4(0.f, 0.f, 0.f, 0.f);
    if (gh >= 0 && gh < H_IMG && gw >= 0 && gw < W_IMG) {
      const float* src =
          x + ((((long)b * H_IMG + gh) * W_IMG + gw) * CI + ci) * AI + ai4 * 4;
      v = *(const float4*)src;
    }
    union { _Float16 hh[4]; uint2 u; } pk;
    pk.hh[0] = (_Float16)v.x; pk.hh[1] = (_Float16)v.y;
    pk.hh[2] = (_Float16)v.z; pk.hh[3] = (_Float16)v.w;
    *(uint2*)&patch[((ci * 5 + r) * 20 + c) * PSTR + ai4 * 4] = pk.u;
  }
  __syncthreads();

  // ---- implicit GEMM: M=16 px, N=16 ch (per wave), K=800 in 32-chunks ----
  const int Nl = lane & 15;   // A row M and B column N for this lane
  const int hi = lane >> 4;   // half-wave selector per ISA fragment layout

  const _Float16* apbase = patch + Nl * PSTR + hi * 16;
  const _Float16* bbase  = Wh + (wv * 16 + Nl) * KTOT + hi * 16;
  const unsigned  abase  = (unsigned)(unsigned long long)apbase;  // LDS addr

  v8f acc0 = {}, acc1 = {}, acc2 = {}, acc3 = {};
  v16h Bring[4];
  v16h A0, A1, A2, A3;

  // B pipeline prologue: 3 chunks (12 WMMAs) of prefetch distance
  glb_load_frag<0 * 64>(Bring[0], bbase);
  glb_load_frag<1 * 64>(Bring[1], bbase);
  glb_load_frag<2 * 64>(Bring[2], bbase);

  CAPS_STEP(0);  CAPS_STEP(1);  CAPS_STEP(2);  CAPS_STEP(3);  CAPS_STEP(4);
  CAPS_STEP(5);  CAPS_STEP(6);  CAPS_STEP(7);  CAPS_STEP(8);  CAPS_STEP(9);
  CAPS_STEP(10); CAPS_STEP(11); CAPS_STEP(12); CAPS_STEP(13); CAPS_STEP(14);
  CAPS_STEP(15); CAPS_STEP(16); CAPS_STEP(17); CAPS_STEP(18); CAPS_STEP(19);
  CAPS_STEP(20); CAPS_STEP(21); CAPS_STEP(22); CAPS_STEP(23); CAPS_STEP(24);

  // ---- scatter votes to LDS as float4 per (pixel, n): [px][256][ci] ----
  float4* v4 = (float4*)vlds;
#pragma unroll
  for (int r = 0; r < 8; ++r) {
    int M = r + hi * 8;                       // D-fragment row per ISA layout
    v4[M * NCH + wv * 16 + Nl] =
        make_float4(acc0[r], acc1[r], acc2[r], acc3[r]);
  }
  __syncthreads();

  // ---- dynamic routing (3 iters), wave v handles pixel v ----
  const int a   = lane;        // atom lane role
  const int ciL = lane >> 3;   // (ci,co) lane role for softmax/agreement
  const int coL = lane & 7;
  float logit = 0.f;
  float bv[CO];
#pragma unroll
  for (int co = 0; co < CO; ++co) bv[co] = bias[co * AT + a];

  const float4* vmy4 = (const float4*)vlds + wv * NCH;  // my pixel's votes
  const float*  vmyf = (const float*)vmy4;

  for (int it = 0; it < 3; ++it) {
    // softmax over co inside 8-lane groups (lane = ci*8+co)
    float m = logit;
    m = fmaxf(m, __shfl_xor(m, 1, 32));
    m = fmaxf(m, __shfl_xor(m, 2, 32));
    m = fmaxf(m, __shfl_xor(m, 4, 32));
    float e = __expf(logit - m);
    float s = e;
    s += __shfl_xor(s, 1, 32);
    s += __shfl_xor(s, 2, 32);
    s += __shfl_xor(s, 4, 32);
    rlds[wv * 32 + lane] = e / s;
    __syncthreads();

    // preact = sum_ci route*votes + bias; squash over the 32 atoms (= lanes)
    float act[CO];
#pragma unroll
    for (int co = 0; co < CO; ++co) {
      float4 vv = vmy4[co * AT + a];          // votes for all 4 ci at (co,a)
      float r0 = rlds[wv * 32 + 0 * 8 + co];
      float r1 = rlds[wv * 32 + 1 * 8 + co];
      float r2 = rlds[wv * 32 + 2 * 8 + co];
      float r3 = rlds[wv * 32 + 3 * 8 + co];
      float p = bv[co] + r0 * vv.x + r1 * vv.y + r2 * vv.z + r3 * vv.w;
      float nsq = p * p;
      nsq += __shfl_xor(nsq, 1, 32);
      nsq += __shfl_xor(nsq, 2, 32);
      nsq += __shfl_xor(nsq, 4, 32);
      nsq += __shfl_xor(nsq, 8, 32);
      nsq += __shfl_xor(nsq, 16, 32);
      float scale = (nsq / (1.0f + nsq)) * rsqrtf(nsq);
      act[co] = p * scale;
    }

    if (it == 2) {
      float* dst = out + (((long)b * H_IMG + h) * W_IMG + (w0 + wv)) * (CO * AT);
#pragma unroll
      for (int co = 0; co < CO; ++co) dst[co * AT + a] = act[co];
    } else {
#pragma unroll
      for (int co = 0; co < CO; ++co) alds[wv * NCH + co * AT + a] = act[co];
      __syncthreads();
      // agreement: logits[ci][co] += <votes[ci][co][:], act[co][:]>
      float agg = 0.f;
#pragma unroll 8
      for (int a2 = 0; a2 < AT; ++a2)
        agg += vmyf[(coL * AT + a2) * CI + ciL] * alds[wv * NCH + coL * AT + a2];
      logit += agg;
      __syncthreads();
    }
  }
}

extern "C" void kernel_launch(void* const* d_in, const int* in_sizes, int n_in,
                              void* d_out, int out_size, void* d_ws, size_t ws_size,
                              hipStream_t stream) {
  const float* x    = (const float*)d_in[0];  // [4,128,128,4,32] f32
  const float* W    = (const float*)d_in[1];  // [5,5,32,256]     f32
  const float* bias = (const float*)d_in[2];  // [1,1,8,32]       f32
  float* out = (float*)d_out;                 // [4,128,128,8,32] f32
  _Float16* Wh = (_Float16*)d_ws;             // needs 409,600 B

  (void)in_sizes; (void)n_in; (void)out_size; (void)ws_size;

  // allow >64KB dynamic LDS (320KB/WGP on gfx1250)
  hipFuncSetAttribute((const void*)capsconv_fused_kernel,
                      hipFuncAttributeMaxDynamicSharedMemorySize, SMEM_BYTES);

  repack_w_kernel<<<(NCH * KTOT + 255) / 256, 256, 0, stream>>>(W, Wh);

  dim3 grid(W_IMG / 16, H_IMG, B_SZ);   // 8 x 128 x 4 = 4096 blocks
  capsconv_fused_kernel<<<grid, 512, SMEM_BYTES, stream>>>(x, Wh, bias, out);
}
